// MultiheadAttention_22436909154459
// MI455X (gfx1250) — compile-verified, matrix operand order checked by reading, not run on hardware
//
#include <hip/hip_runtime.h>
#include <hip/hip_bf16.h>

typedef __attribute__((ext_vector_type(16))) __bf16 v16bf;
typedef __attribute__((ext_vector_type(8)))  __bf16 v8bf;
typedef __attribute__((ext_vector_type(4)))  __bf16 v4bf;
typedef __attribute__((ext_vector_type(8)))  float  v8f;
typedef __attribute__((ext_vector_type(4)))  unsigned int u32x4;
typedef __attribute__((ext_vector_type(4)))  int i32x4;
typedef __attribute__((ext_vector_type(8)))  int i32x8;

// Problem constants (reference: T=S=1024, N=8, E=1024, H=16, D=64)
#define TT   1024
#define SS   1024
#define NB   8
#define EE   1024
#define HH   16
#define DD   64
#define MROWS (TT * NB)      // 8192
#define F3E  (3 * EE)        // 3072

#if defined(__has_builtin)
#if __has_builtin(__builtin_amdgcn_tensor_load_to_lds) && __has_builtin(__builtin_amdgcn_s_wait_tensorcnt)
#define HAVE_TDM 1
#endif
#endif
#ifndef HAVE_TDM
#define HAVE_TDM 0
#endif

// ---------------------------------------------------------------------------
// TDM 2D tile load: global (rows x cols bf16, row stride ld elems) -> LDS
// (packed row-major rows x cols). Issued by wave 0 only (EXEC is ignored by
// TENSOR ops; one issue per workgroup). Completion via s_wait_tensorcnt.
// D# layout per CDNA5 ISA.
// ---------------------------------------------------------------------------
__device__ __forceinline__ void tile_load(int wave, int tid, __bf16* dst,
                                          const __bf16* src, int rows, int cols, int ld) {
#if HAVE_TDM
    if (wave == 0) {
        unsigned long long ga = (unsigned long long)(size_t)src;
        unsigned lds_off = (unsigned)(size_t)dst;   // flat LDS addr low 32 bits = LDS byte offset
        u32x4 g0;
        g0[0] = 1u;                                              // count=1, user descriptor
        g0[1] = lds_off;                                         // D#.lds_addr
        g0[2] = (unsigned)ga;                                    // global_addr[31:0]
        g0[3] = (unsigned)((ga >> 32) & 0x01FFFFFFull) | (2u << 30);  // global_addr[56:32] | type=2
        i32x8 g1;
        g1[0] = (1 << 16);                                       // data_size=1 (2 bytes/elem)
        g1[1] = (int)(((unsigned)cols & 0xFFFFu) << 16);         // tensor_dim0[15:0]
        g1[2] = (int)((((unsigned)cols >> 16) & 0xFFFFu) |
                      (((unsigned)rows & 0xFFFFu) << 16));       // tensor_dim0[31:16] | tensor_dim1[15:0]
        g1[3] = (int)((((unsigned)rows >> 16) & 0xFFFFu) |
                      (((unsigned)cols & 0xFFFFu) << 16));       // tensor_dim1[31:16] | tile_dim0
        g1[4] = rows;                                            // tile_dim1 (tile_dim2 = 0)
        g1[5] = ld;                                              // tensor_dim0_stride[31:0] (elems)
        g1[6] = 0;                                               // stride0 hi | stride1 lo
        g1[7] = 0;                                               // stride1 hi
        i32x4 z4 = {0, 0, 0, 0};
#if __clang_major__ >= 23
        i32x8 z8 = {0, 0, 0, 0, 0, 0, 0, 0};
        __builtin_amdgcn_tensor_load_to_lds(g0, g1, z4, z4, z8, 0);
#else
        __builtin_amdgcn_tensor_load_to_lds(g0, g1, z4, z4, 0);
#endif
    }
#else
    // Fallback: cooperative vectorized copy (cols is a multiple of 8)
    const int total = rows * cols;
    for (int idx = tid * 8; idx < total; idx += 256 * 8) {
        int r = idx / cols, c = idx % cols;
        *(v8bf*)&dst[idx] = *(const v8bf*)&src[(size_t)r * ld + c];
    }
#endif
    (void)tid; (void)wave;
}

__device__ __forceinline__ void tile_wait(int wave) {
#if HAVE_TDM
    if (wave == 0) __builtin_amdgcn_s_wait_tensorcnt(0);
#endif
    (void)wave;
}

// ---------------------------------------------------------------------------
// Fragment loader: storage is packed row-major [row][K] bf16; WMMA A/B
// fragment element j of lane maps to K = (j>>3)*16 + (lane>>4)*8 + (j&7),
// row = lane&15  => two contiguous 16B LDS loads per fragment.
// ---------------------------------------------------------------------------
__device__ __forceinline__ v16bf frag_ld(const __bf16* base, int ld,
                                         int row_base, int k_base, int lane) {
    const __bf16* p = base + (row_base + (lane & 15)) * ld + k_base + ((lane >> 4) << 3);
    v8bf lo = *(const v8bf*)(p);
    v8bf hi = *(const v8bf*)(p + 16);
    return __builtin_shufflevector(lo, hi, 0,1,2,3,4,5,6,7,8,9,10,11,12,13,14,15);
}

__device__ __forceinline__ v8f wmma_bf16(v16bf a, v16bf b, v8f c) {
    return __builtin_amdgcn_wmma_f32_16x16x32_bf16(
        false, a, false, b, (short)0, c, false, false);
}

// ---------------------------------------------------------------------------
// fp32 -> bf16 cast kernel (4 elems/thread)
// ---------------------------------------------------------------------------
__global__ __launch_bounds__(256)
void cvt_f32_to_bf16(const float* __restrict__ in, __bf16* __restrict__ out, int n) {
    int i = (blockIdx.x * 256 + threadIdx.x) * 4;
    if (i < n) {
        float4 v = *(const float4*)(in + i);
        v4bf o; o[0] = (__bf16)v.x; o[1] = (__bf16)v.y; o[2] = (__bf16)v.z; o[3] = (__bf16)v.w;
        *(v4bf*)(out + i) = o;
    }
}

// ---------------------------------------------------------------------------
// bf16 GEMM:  C[M,N] = A[M,K] * B[N,K]^T + bias[N]
// mode 0: out = bf16 (qkv projection, q-columns (n<1024) scaled by D^-0.5)
// mode 1: out = f32  (final out-projection into d_out)
// 128x128x32 block tile, 8 waves x (4x2) 16x16 WMMA tiles.
// TDM double-buffered LDS slabs: DMA of slab k+1 overlaps WMMAs on slab k.
// ---------------------------------------------------------------------------
#define BM 128
#define BN 128
#define BK 32

__global__ __launch_bounds__(256)
void gemm_bf16(const __bf16* __restrict__ A, const __bf16* __restrict__ B,
               const float* __restrict__ bias,
               __bf16* __restrict__ outBf, float* __restrict__ outF32,
               int M, int N, int K, int mode) {
    __shared__ __bf16 Asl[2][BM * BK];
    __shared__ __bf16 Bsl[2][BM * BK];

    const int tid  = threadIdx.x;
    const int lane = tid & 31;
    const int wave = tid >> 5;
    const int m0 = blockIdx.y * BM;
    const int n0 = blockIdx.x * BN;
    const int wm = (wave >> 2) * 64;
    const int wn = (wave & 3) * 32;

    v8f acc[4][2] = {};

    const __bf16* Ab = A + (size_t)m0 * K;
    const __bf16* Bb = B + (size_t)n0 * K;
    const int nk = K / BK;

    tile_load(wave, tid, Asl[0], Ab, BM, BK, K);
    tile_load(wave, tid, Bsl[0], Bb, BN, BK, K);

    for (int ki = 0; ki < nk; ++ki) {
        tile_wait(wave);
        __syncthreads();                    // current slab resident; prev reads done
        if (ki + 1 < nk) {                  // overlap next slab DMA with compute
            tile_load(wave, tid, Asl[(ki + 1) & 1], Ab + (size_t)(ki + 1) * BK, BM, BK, K);
            tile_load(wave, tid, Bsl[(ki + 1) & 1], Bb + (size_t)(ki + 1) * BK, BN, BK, K);
        }
        const __bf16* As = Asl[ki & 1];
        const __bf16* Bs = Bsl[ki & 1];

        v16bf af[4], bfg[2];
#pragma unroll
        for (int i = 0; i < 4; ++i) af[i]  = frag_ld(As, BK, wm + i * 16, 0, lane);
#pragma unroll
        for (int j = 0; j < 2; ++j) bfg[j] = frag_ld(Bs, BK, wn + j * 16, 0, lane);
#pragma unroll
        for (int i = 0; i < 4; ++i)
#pragma unroll
            for (int j = 0; j < 2; ++j)
                acc[i][j] = wmma_bf16(af[i], bfg[j], acc[i][j]);
    }

    // Epilogue: C layout — VGPR r: M = (lane>>4)*8 + r, N = lane&15
#pragma unroll
    for (int i = 0; i < 4; ++i)
#pragma unroll
        for (int j = 0; j < 2; ++j)
#pragma unroll
            for (int r = 0; r < 8; ++r) {
                int gm = m0 + wm + i * 16 + ((lane >> 4) << 3) + r;
                int gn = n0 + wn + j * 16 + (lane & 15);
                float v = acc[i][j][r] + bias[gn];
                if (mode == 0) {
                    if (gn < EE) v *= 0.125f;        // q * D^-0.5, D = 64
                    outBf[(size_t)gm * N + gn] = (__bf16)v;
                } else {
                    outF32[(size_t)gm * N + gn] = v;
                }
            }
}

// ---------------------------------------------------------------------------
// Fused attention: block = (32-row T-tile, n); loops all 16 heads with the
// 32x1024 scores block and head-averaged probs resident in LDS. K chunks are
// TDM double-buffered; V chunks are transposed through VGPRs (TDM has no
// transpose). qkv (48 MB) stays L2-resident across re-reads.
// ---------------------------------------------------------------------------
#define ATTN_SMEM_BYTES (32*1024*4 + 32*1024*4 + 32*64*2 + 2*64*64*2)  // 282624

__global__ __launch_bounds__(256)
void attn_fused(const __bf16* __restrict__ qkv, __bf16* __restrict__ ctx,
                float* __restrict__ avg_out) {
    extern __shared__ char smem[];
    float*  sc  = (float*)smem;               // 32 x 1024 scores / probs
    float*  av  = sc + 32 * 1024;             // 32 x 1024 head-averaged probs
    __bf16* qt  = (__bf16*)(av + 32 * 1024);  // 32 x 64 Q tile
    __bf16* kb0 = qt + 32 * 64;               // 64 x 64 K/V chunk buf 0
    __bf16* kb1 = kb0 + 64 * 64;              // 64 x 64 K chunk buf 1

    const int n    = blockIdx.y;
    const int t0   = blockIdx.x * 32;
    const int tid  = threadIdx.x;
    const int lane = tid & 31;
    const int wave = tid >> 5;
    const int tm   = wave & 1;   // M-subtile (2 x 16 rows)
    const int tq   = wave >> 1;  // N-subtile (4 x 16 cols)
    const int LD   = NB * F3E;   // row stride (elems) between consecutive t/s rows

    for (int i = tid; i < 32 * 1024; i += 256) av[i] = 0.0f;

    for (int h = 0; h < HH; ++h) {
        const __bf16* Qhead = qkv + (size_t)t0 * LD + (size_t)n * F3E + h * DD;
        const __bf16* Khead = qkv + (size_t)n * F3E + EE + h * DD;
        const __bf16* Vhead = qkv + (size_t)n * F3E + 2 * EE + h * DD;

        // ---- issue Q tile + first K chunk DMA ----
        tile_load(wave, tid, qt,  Qhead, 32, 64, LD);
        tile_load(wave, tid, kb0, Khead, 64, 64, LD);

        // ---- scores = Q K^T over S, 64-column chunks (double-buffered K) ----
        for (int c = 0; c < SS / 64; ++c) {
            tile_wait(wave);
            __syncthreads();
            if (c + 1 < SS / 64)
                tile_load(wave, tid, ((c + 1) & 1) ? kb1 : kb0,
                          Khead + (size_t)(c + 1) * 64 * LD, 64, 64, LD);
            const __bf16* Ks = (c & 1) ? kb1 : kb0;

            v8f cacc = {};
#pragma unroll
            for (int kk = 0; kk < 64; kk += 32) {
                v16bf a = frag_ld(qt, 64, tm * 16, kk, lane);  // A: M=t, K=d
                v16bf b = frag_ld(Ks, 64, tq * 16, kk, lane);  // B: N=s, K=d
                cacc = wmma_bf16(a, b, cacc);
            }
#pragma unroll
            for (int r = 0; r < 8; ++r) {
                int rr = tm * 16 + ((lane >> 4) << 3) + r;
                int cc = c * 64 + tq * 16 + (lane & 15);
                sc[rr * 1024 + cc] = cacc[r];
            }
        }
        __syncthreads();

        // ---- softmax over S=1024 (wave w owns rows 4w..4w+3); fold into avg ----
        for (int i = 0; i < 4; ++i) {
            int row = wave * 4 + i;
            float* pr = &sc[row * 1024];
            float m = -1e30f;
            for (int c2 = lane; c2 < 1024; c2 += 32) m = fmaxf(m, pr[c2]);
#pragma unroll
            for (int o = 16; o >= 1; o >>= 1) m = fmaxf(m, __shfl_xor(m, o, 32));
            float s = 0.0f;
            for (int c2 = lane; c2 < 1024; c2 += 32) {
                float e = __expf(pr[c2] - m);
                pr[c2] = e;
                s += e;
            }
#pragma unroll
            for (int o = 16; o >= 1; o >>= 1) s += __shfl_xor(s, o, 32);
            float inv = 1.0f / s;
            for (int c2 = lane; c2 < 1024; c2 += 32) {
                float p = pr[c2] * inv;
                pr[c2] = p;
                av[row * 1024 + c2] += p * (1.0f / (float)HH);
            }
        }
        __syncthreads();

        // ---- O = P V : accumulate over S in 64-chunks (V staged transposed) ----
        v8f o = {};
        for (int s0 = 0; s0 < SS; s0 += 64) {
            int srow = tid >> 2, dseg = (tid & 3) << 4;
            const __bf16* vsrc = Vhead + (size_t)(s0 + srow) * LD + dseg;
            v8bf v0 = *(const v8bf*)(vsrc);
            v8bf v1 = *(const v8bf*)(vsrc + 8);
#pragma unroll
            for (int d2 = 0; d2 < 8; ++d2) {
                kb0[(dseg + d2) * 64 + srow]     = v0[d2];
                kb0[(dseg + 8 + d2) * 64 + srow] = v1[d2];
            }
            __syncthreads();

#pragma unroll
            for (int kk = 0; kk < 64; kk += 32) {
                const float* ap = &sc[(tm * 16 + (lane & 15)) * 1024 + s0 + kk + ((lane >> 4) << 3)];
                v16bf a;
#pragma unroll
                for (int j = 0; j < 8; ++j) {
                    a[j]     = (__bf16)ap[j];
                    a[j + 8] = (__bf16)ap[16 + j];
                }
                v16bf b = frag_ld(kb0, 64, tq * 16, kk, lane);  // B: N=d, K=s
                o = wmma_bf16(a, b, o);
            }
            __syncthreads();
        }

        // ---- store context tile (bf16) for the out-projection GEMM ----
#pragma unroll
        for (int r = 0; r < 8; ++r) {
            int gt = t0 + tm * 16 + ((lane >> 4) << 3) + r;
            int gd = tq * 16 + (lane & 15);
            ctx[((size_t)gt * NB + n) * EE + h * DD + gd] = (__bf16)o[r];
        }
        __syncthreads();
    }

    // ---- write avg_weights (N,T,S) ----
    for (int i = tid; i < 32 * 1024; i += 256) {
        int row = i >> 10, col = i & 1023;
        avg_out[((size_t)n * TT + (t0 + row)) * SS + col] = av[i];
    }
}

// ---------------------------------------------------------------------------
// Host launcher
// ---------------------------------------------------------------------------
extern "C" void kernel_launch(void* const* d_in, const int* in_sizes, int n_in,
                              void* d_out, int out_size, void* d_ws, size_t ws_size,
                              hipStream_t stream) {
    const float* query = (const float*)d_in[0];
    // d_in[1] (key) and d_in[2] (value) are unused by the reference
    const float* w_in  = (const float*)d_in[3];
    const float* b_in  = (const float*)d_in[4];
    const float* w_out = (const float*)d_in[5];
    const float* b_out = (const float*)d_in[6];

    __bf16* q_bf    = (__bf16*)d_ws;                               // 8192*1024
    __bf16* win_bf  = q_bf   + (size_t)MROWS * EE;                 // 3072*1024
    __bf16* wout_bf = win_bf + (size_t)F3E * EE;                   // 1024*1024
    __bf16* qkv     = wout_bf + (size_t)EE * EE;                   // 8192*3072
    __bf16* ctx     = qkv    + (size_t)MROWS * F3E;                // 8192*1024

    float* outp = (float*)d_out;                                   // (T,N,E)
    float* avg  = outp + (size_t)MROWS * EE;                       // (N,T,S)

    // 1) fp32 -> bf16 casts
    {
        int n1 = MROWS * EE, n2 = F3E * EE, n3 = EE * EE;
        cvt_f32_to_bf16<<<(n1 / 4 + 255) / 256, 256, 0, stream>>>(query, q_bf, n1);
        cvt_f32_to_bf16<<<(n2 / 4 + 255) / 256, 256, 0, stream>>>(w_in, win_bf, n2);
        cvt_f32_to_bf16<<<(n3 / 4 + 255) / 256, 256, 0, stream>>>(w_out, wout_bf, n3);
    }

    // 2) QKV projection (q pre-scaled by D^-0.5), bf16 output
    {
        dim3 g(F3E / BN, MROWS / BM);  // (24, 64)
        gemm_bf16<<<g, 256, 0, stream>>>(q_bf, win_bf, b_in, qkv, nullptr,
                                         MROWS, F3E, EE, 0);
    }

    // 3) Fused attention + head-averaged weights
    {
        hipFuncSetAttribute((const void*)attn_fused,
                            hipFuncAttributeMaxDynamicSharedMemorySize,
                            ATTN_SMEM_BYTES);
        dim3 g(TT / 32, NB);           // (32, 8)
        attn_fused<<<g, 256, ATTN_SMEM_BYTES, stream>>>(qkv, ctx, avg);
    }

    // 4) Output projection, f32 output straight into d_out
    {
        dim3 g(EE / BN, MROWS / BM);   // (8, 64)
        gemm_bf16<<<g, 256, 0, stream>>>(ctx, wout_bf, b_out, nullptr, outp,
                                         MROWS, EE, EE, 1);
    }
}
